// ModelWithLoss_67808943669610
// MI455X (gfx1250) — matrix operand order repaired and checked by reading, not compiled, not run on hardware
//
#include <hip/hip_runtime.h>
#include <hip/hip_bf16.h>

// ---------------------------------------------------------------------------
// Problem constants (match the reference)
// ---------------------------------------------------------------------------
constexpr int Bn = 1024;    // batch
constexpr int Dn = 64;      // feature dim (K of the GEMM)
constexpr int Cn = 100000;  // classes (N of the GEMM)
constexpr int Kp = 5;       // positives per sample

// Tiling
constexpr int CPW   = 512;                      // columns per workgroup
constexpr int CHUNK = 128;                      // columns staged in LDS at a time
constexpr int NCH   = CPW / CHUNK;              // 4
constexpr int NW    = (Cn + CPW - 1) / CPW;     // 196 workgroups
constexpr int XSTR  = 72;                       // LDS row stride (halfs), 16B-aligned, bank-spread
constexpr int WSTR  = 72;

typedef _Float16 h8  __attribute__((ext_vector_type(8)));
typedef _Float16 h16 __attribute__((ext_vector_type(16)));
typedef float    f8  __attribute__((ext_vector_type(8)));

union H16u { h16 v; h8 p[2]; };

// ---------------------------------------------------------------------------
// Kernel 1: fused f16-WMMA GEMM (x@W + b) with online per-row softmax stats.
// Block = 256 threads = 8 waves. Block `wg` owns columns [wg*CPW, wg*CPW+CPW).
// Round r stages x rows [r*128, r*128+128) to LDS (f16); wave w owns M-tile
// r*8+w (16 rows) and sweeps all CPW columns, keeping per-lane running
// (max, sum-exp). Cross-lane merge once per tile, partials -> workspace.
// Bias for the chunk is staged to LDS (with -1e30 padding) so the inner
// WMMA loop has no divergent global accesses and EXEC stays all-ones.
// ---------------------------------------------------------------------------
__global__ __launch_bounds__(256) void gemm_softmax_partial(
    const float* __restrict__ x, const float* __restrict__ Wm,
    const float* __restrict__ bias,
    float* __restrict__ wsm, float* __restrict__ wss)
{
    __shared__ __align__(16) _Float16 xs[128 * XSTR];     // 18 KB
    __shared__ __align__(16) _Float16 wt[CHUNK * WSTR];   // 18 KB (transposed W chunk)
    __shared__ __align__(16) float    bsh[CHUNK];         // 512 B (bias, padded)

    const int tid = threadIdx.x, wg = blockIdx.x;
    const int wv = tid >> 5, lane = tid & 31;
    const int Lm = lane & 15, hf = lane >> 4;
    const int cbase = wg * CPW;

    for (int r = 0; r < 8; ++r) {
        __syncthreads();  // WAR on xs from previous round
        // ---- stage 128 rows of x as f16 (coalesced global reads) ----
        for (int idx = tid; idx < 128 * Dn; idx += 256) {
            const int row = idx >> 6, k = idx & 63;
            xs[row * XSTR + k] = (_Float16)x[(r * 128 + row) * Dn + k];
        }
        __syncthreads();

        // ---- A operands for this wave's 16-row tile (held across chunks) ----
        // 16-bit A 16x32 layout: lane<16 holds M=Lm, K {0..7,16..23}; lane>=16 K {8..15,24..31}
        H16u A1, A2;
        const _Float16* ap = xs + (wv * 16 + Lm) * XSTR + hf * 8;
        A1.p[0] = *(const h8*)(ap);        // K 0..7   / 8..15
        A1.p[1] = *(const h8*)(ap + 16);   // K 16..23 / 24..31
        A2.p[0] = *(const h8*)(ap + 32);   // K 32..39 / 40..47
        A2.p[1] = *(const h8*)(ap + 48);   // K 48..55 / 56..63

        float m[8], s[8];
#pragma unroll
        for (int i = 0; i < 8; ++i) { m[i] = -__builtin_inff(); s[i] = 0.f; }

        for (int ch = 0; ch < NCH; ++ch) {
            __syncthreads();  // WAR on wt/bsh
            const int cb = cbase + ch * CHUNK;
            // ---- stage W chunk: coalesced f32 read, f16 transposed store ----
            for (int idx = tid; idx < CHUNK * Dn; idx += 256) {
                const int d = idx >> 7, cc = idx & 127;
                const int cg = cb + cc;
                const float v = (cg < Cn) ? Wm[d * Cn + cg] : 0.f;
                wt[cc * WSTR + d] = (_Float16)v;
                if ((ch + 1 < NCH) && ((idx & 127) == 0))
                    __builtin_prefetch(&Wm[d * Cn + cb + CHUNK], 0, 1);  // global_prefetch_b8
            }
            // ---- stage bias for the chunk (padding -> ~-inf) ----
            if (tid < CHUNK) {
                const int cg = cb + tid;
                bsh[tid] = (cg < Cn) ? bias[cg] : -1e30f;
            }
            __syncthreads();

#pragma unroll
            for (int ct = 0; ct < 8; ++ct) {
                // 16-bit B 32x16 layout: lane holds N=Lm, K {hf*8+0..7, 16+hf*8+0..7}
                const _Float16* bp = wt + (ct * 16 + Lm) * WSTR + hf * 8;
                H16u B1, B2;
                B1.p[0] = *(const h8*)(bp);       B1.p[1] = *(const h8*)(bp + 16);
                B2.p[0] = *(const h8*)(bp + 32);  B2.p[1] = *(const h8*)(bp + 48);

                f8 c = {0.f, 0.f, 0.f, 0.f, 0.f, 0.f, 0.f, 0.f};
                c = __builtin_amdgcn_wmma_f32_16x16x32_f16(
                        false, A1.v, false, B1.v, (short)0, c, false, false);
                c = __builtin_amdgcn_wmma_f32_16x16x32_f16(
                        false, A2.v, false, B2.v, (short)0, c, false, false);

                const float bc = bsh[ct * 16 + Lm];
#pragma unroll
                for (int i = 0; i < 8; ++i) {
                    // online softmax, one v_exp per logit, branchless (EXEC stays full)
                    const float l  = c[i] + bc;
                    const float d1 = l - m[i];
                    const float e  = __expf(-fabsf(d1));
                    const bool gt  = d1 > 0.f;
                    s[i] = gt ? fmaf(s[i], e, 1.f) : (s[i] + e);
                    m[i] = gt ? l : m[i];
                }
            }
        }

        // ---- merge (m,s) across the 16 lanes of each half-wave ----
#pragma unroll
        for (int mask = 1; mask < 16; mask <<= 1) {
#pragma unroll
            for (int i = 0; i < 8; ++i) {
                const float mo = __shfl_xor(m[i], mask, 32);
                const float so = __shfl_xor(s[i], mask, 32);
                const float mn = fmaxf(m[i], mo);
                s[i] = s[i] * __expf(m[i] - mn) + so * __expf(mo - mn);
                m[i] = mn;
            }
        }
        if (Lm == 0) {  // lane 0 -> rows m0..m0+7, lane 16 -> rows m0+8..m0+15
            const int m0 = (r * 8 + wv) * 16 + hf * 8;
#pragma unroll
            for (int i = 0; i < 8; ++i) {
                wsm[(m0 + i) * NW + wg] = m[i];
                wss[(m0 + i) * NW + wg] = s[i];
            }
        }
    }
}

// ---------------------------------------------------------------------------
// Kernel 2: pos_logits[b,k] = dot(x[b,:], W[:,label]) + bias[label]  (fp32)
// ---------------------------------------------------------------------------
__global__ __launch_bounds__(256) void pos_logits_kernel(
    const float* __restrict__ x, const int* __restrict__ labels,
    const float* __restrict__ Wm, const float* __restrict__ bias,
    float* __restrict__ wsp)
{
    const int t = blockIdx.x * blockDim.x + threadIdx.x;
    if (t >= Bn * Kp) return;
    const int bi  = t / Kp;
    const int lab = labels[t];
    float acc = bias[lab];
#pragma unroll
    for (int d = 0; d < Dn; ++d)
        acc = fmaf(x[bi * Dn + d], Wm[d * Cn + lab], acc);
    wsp[t] = acc;
}

// ---------------------------------------------------------------------------
// Kernel 3: per-row merge of NW partials, loss, mean via atomicAdd.
// ---------------------------------------------------------------------------
__global__ __launch_bounds__(256) void reduce_loss_kernel(
    const float* __restrict__ wsm, const float* __restrict__ wss,
    const float* __restrict__ wsp, float* __restrict__ out)
{
    const int b = blockIdx.x * blockDim.x + threadIdx.x;
    if (b >= Bn) return;
    float m = -__builtin_inff();
    for (int j = 0; j < NW; ++j) m = fmaxf(m, wsm[b * NW + j]);
    float tot = 0.f;
    for (int j = 0; j < NW; ++j)
        tot += wss[b * NW + j] * __expf(wsm[b * NW + j] - m);

    float pl[Kp], pe[Kp], psum = 0.f;
#pragma unroll
    for (int k = 0; k < Kp; ++k) {
        pl[k] = wsp[b * Kp + k];
        pe[k] = __expf(pl[k] - m);
        psum += pe[k];
    }
    const float neg = tot - psum;
    float acc = 0.f;
#pragma unroll
    for (int k = 0; k < Kp; ++k)
        acc += -(pl[k] - m - __logf(pe[k] + neg));
    atomicAdd(out, acc * (1.f / (float)(Bn * Kp)));
}

__global__ void zero_out_kernel(float* out) { out[0] = 0.f; }

// ---------------------------------------------------------------------------
// Launch
// ---------------------------------------------------------------------------
extern "C" void kernel_launch(void* const* d_in, const int* in_sizes, int n_in,
                              void* d_out, int out_size, void* d_ws, size_t ws_size,
                              hipStream_t stream)
{
    (void)in_sizes; (void)n_in; (void)out_size; (void)ws_size;
    const float* x      = (const float*)d_in[0];   // [B, D]
    const int*   labels = (const int*)d_in[1];     // [B, K]
    const float* Wm     = (const float*)d_in[2];   // [D, C]
    const float* bias   = (const float*)d_in[3];   // [C]
    float* out = (float*)d_out;

    float* wsf = (float*)d_ws;
    float* wsm = wsf;                        // [B, NW] chunk maxes
    float* wss = wsf + (size_t)Bn * NW;      // [B, NW] chunk exp-sums
    float* wsp = wsf + (size_t)2 * Bn * NW;  // [B, K] positive logits

    zero_out_kernel<<<1, 1, 0, stream>>>(out);
    gemm_softmax_partial<<<dim3(NW), dim3(256), 0, stream>>>(x, Wm, bias, wsm, wss);
    pos_logits_kernel<<<dim3((Bn * Kp + 255) / 256), dim3(256), 0, stream>>>(
        x, labels, Wm, bias, wsp);
    reduce_loss_kernel<<<dim3(Bn / 256), dim3(256), 0, stream>>>(wsm, wss, wsp, out);
}
